// CrossAttention_22497038696777
// MI455X (gfx1250) — matrix operand order repaired and checked by reading, not compiled, not run on hardware
//
#include <hip/hip_runtime.h>
#include <hip/hip_bf16.h>

typedef __attribute__((ext_vector_type(16))) _Float16 v16h;
typedef __attribute__((ext_vector_type(8)))  _Float16 v8h;
typedef __attribute__((ext_vector_type(8)))  float    v8f;

#define B_     2
#define N_     2048
#define M_     2048
#define HEADS_ 16
#define DH_    64
#define INNER_ 1024

// ---------------------------------------------------------------------------
// WMMA helpers
// ---------------------------------------------------------------------------
__device__ __forceinline__ v8f wmma_f16(v16h a, v16h b, v8f c) {
  // D = A(16x32 f16) * B(32x16 f16) + C(16x16 f32)
  return __builtin_amdgcn_wmma_f32_16x16x32_f16(
      /*neg_a=*/false, a, /*neg_b=*/false, b,
      /*c_mod=*/(short)0, c, /*reuse_a=*/false, /*reuse_b=*/false);
}

// A-fragment (16x32, f16): lane holds row M=lane%16; K elements split
// {0..7,16..23} (lanes 0-15) / {8..15,24..31} (lanes 16-31).
__device__ __forceinline__ v16h load_fragA(const _Float16* base, int ld) {
  const int lane = threadIdx.x & 31;
  const int r    = lane & 15;
  const int kb   = (lane >> 4) << 3;         // 0 or 8
  const _Float16* p = base + r * ld + kb;
  v8h lo = *(const v8h*)(p);                 // K = kb .. kb+7
  v8h hi = *(const v8h*)(p + 16);            // K = kb+16 .. kb+23
  v16h f;
#pragma unroll
  for (int i = 0; i < 8; ++i) { f[i] = lo[i]; f[i + 8] = hi[i]; }
  return f;
}

// B-fragment (32x16, f16): lane holds column N=lane%16; K elements split
// {0..7,16..23} (lanes 0-15) / {8..15,24..31} (lanes 16-31). Source stored N-major.
__device__ __forceinline__ v16h load_fragB(const _Float16* base, int ld) {
  const int lane = threadIdx.x & 31;
  const int r    = lane & 15;
  const int kb   = (lane >> 4) << 3;         // 0 or 8
  const _Float16* p = base + r * ld + kb;
  v8h lo = *(const v8h*)(p);
  v8h hi = *(const v8h*)(p + 16);
  v16h f;
#pragma unroll
  for (int i = 0; i < 8; ++i) { f[i] = lo[i]; f[i + 8] = hi[i]; }
  return f;
}

// ---------------------------------------------------------------------------
// f32 -> f16 convert (8 elems/thread) / transpose-convert
// ---------------------------------------------------------------------------
__global__ void cvt_f16_kernel(const float* __restrict__ in,
                               _Float16* __restrict__ out, int n8) {
  int i = blockIdx.x * blockDim.x + threadIdx.x;
  if (i < n8) {
    const float4* p = (const float4*)in + (size_t)i * 2;
    float4 a = p[0], b = p[1];
    v8h o;
    o[0] = (_Float16)a.x; o[1] = (_Float16)a.y;
    o[2] = (_Float16)a.z; o[3] = (_Float16)a.w;
    o[4] = (_Float16)b.x; o[5] = (_Float16)b.y;
    o[6] = (_Float16)b.z; o[7] = (_Float16)b.w;
    ((v8h*)out)[i] = o;
  }
}

// W [K=1024, N=1024] f32 row-major  ->  Wt [N,K] f16 row-major
__global__ void transpose_cvt_kernel(const float* __restrict__ W,
                                     _Float16* __restrict__ Wt) {
  int i = blockIdx.x * blockDim.x + threadIdx.x;   // 0 .. 1M-1
  int k = i >> 10;
  int n = i & 1023;
  Wt[(size_t)n * 1024 + k] = (_Float16)W[(size_t)k * 1024 + n];
}

// ---------------------------------------------------------------------------
// GEMM: C = A[M,K] * W, W supplied transposed (Wt [N,K] row-major).
// Block = 256 threads (8 waves) -> 128x128 tile; wave grid 4x2, each wave
// computes 32x64 (2 A-frags + 4 B-frags per k-step -> 8 WMMAs).
// ---------------------------------------------------------------------------
template <bool F32OUT>
__global__ void __launch_bounds__(256)
gemm_wmma_kernel(const _Float16* __restrict__ A,    // [Mrows, K] row-major
                 const _Float16* __restrict__ Wt,   // [Ncols, K] row-major
                 _Float16* __restrict__ Oh,         // f16 output (or null)
                 float* __restrict__ Of,            // f32 output (or null)
                 const float* __restrict__ bias,    // [Ncols] (or null)
                 int Mrows, int K, int Ncols) {
  const int wave = threadIdx.x >> 5;
  const int lane = threadIdx.x & 31;
  const int row0 = blockIdx.x * 128 + (wave >> 1) * 32;   // 32 rows / wave
  const int col0 = blockIdx.y * 128 + (wave & 1) * 64;    // 64 cols / wave

  v8f acc[2][4];
#pragma unroll
  for (int m = 0; m < 2; ++m)
#pragma unroll
    for (int c = 0; c < 4; ++c) acc[m][c] = (v8f){};

  const _Float16* Aw = A + (size_t)row0 * K;
  const _Float16* Ww = Wt + (size_t)col0 * K;

  for (int k0 = 0; k0 < K; k0 += 32) {
    // Issue all fragment loads before the WMMA group.
    v16h a0 = load_fragA(Aw + k0, K);
    v16h a1 = load_fragA(Aw + (size_t)16 * K + k0, K);
    v16h b[4];
#pragma unroll
    for (int c = 0; c < 4; ++c)
      b[c] = load_fragB(Ww + (size_t)(c * 16) * K + k0, K);
#pragma unroll
    for (int c = 0; c < 4; ++c) {
      acc[0][c] = wmma_f16(a0, b[c], acc[0][c]);
      acc[1][c] = wmma_f16(a1, b[c], acc[1][c]);
    }
  }

  const int hs = lane >> 4;       // C-layout: row r (+8 upper half-wave)
  const int cn = lane & 15;       // col = lane%16
#pragma unroll
  for (int m = 0; m < 2; ++m) {
#pragma unroll
    for (int c = 0; c < 4; ++c) {
#pragma unroll
      for (int r = 0; r < 8; ++r) {
        int row = row0 + m * 16 + r + 8 * hs;
        int col = col0 + c * 16 + cn;
        if (F32OUT)
          Of[(size_t)row * Ncols + col] = acc[m][c][r] + bias[col];
        else
          Oh[(size_t)row * Ncols + col] = (_Float16)acc[m][c][r];
      }
    }
  }
}

// ---------------------------------------------------------------------------
// Flash attention: block = (128 q-rows, one head, one batch); 8 waves,
// each wave owns 16 query rows. Online softmax, key chunks of 64.
// Pipelining: V chunk loaded into regs first, spilled to LDS only after the
// S/softmax phase so the global loads overlap VALU/TRANS work.
// ---------------------------------------------------------------------------
__global__ void __launch_bounds__(256)
attn_wmma_kernel(const _Float16* __restrict__ Qh,
                 const _Float16* __restrict__ Kh,
                 const _Float16* __restrict__ Vh,
                 _Float16* __restrict__ Oh) {
  __shared__ __align__(16) _Float16 sVt[64 * 72];        // V^T chunk [d][key]
  __shared__ __align__(16) _Float16 sP[8][16 * 72];      // per-wave P tile

  const int lane = threadIdx.x & 31;
  const int wave = threadIdx.x >> 5;
  const int qb = blockIdx.x;
  const int h  = blockIdx.y;
  const int b  = blockIdx.z;
  const int qrow0 = qb * 128 + wave * 16;

  const size_t qbase = (size_t)b * N_ * INNER_;
  const size_t kvbase = (size_t)b * M_ * INNER_;
  const _Float16* Qp = Qh + qbase + (size_t)qrow0 * INNER_ + h * DH_;
  const _Float16* Kp = Kh + kvbase + h * DH_;
  const _Float16* Vp = Vh + kvbase + h * DH_;

  // Q fragments (16 rows x 64 d) kept in registers: two K-steps of 32
  v16h qa0 = load_fragA(Qp, INNER_);
  v16h qa1 = load_fragA(Qp + 32, INNER_);

  float mrun[8], lrun[8];
#pragma unroll
  for (int r = 0; r < 8; ++r) { mrun[r] = -1e30f; lrun[r] = 0.f; }
  v8f o[4];
#pragma unroll
  for (int dt = 0; dt < 4; ++dt) o[dt] = (v8f){};

  const float scale = 0.125f;   // 1/sqrt(64)
  const int hs = lane >> 4;
  const int cn = lane & 15;

  for (int kc = 0; kc < M_; kc += 64) {
    // --- Stage 1: issue V chunk loads into registers (b128 each) ----------
    v8h vreg[2];
#pragma unroll
    for (int j = 0; j < 2; ++j) {
      int slot = threadIdx.x + j * 256;       // 512 slots = 64 keys x 8 blk
      int key = slot >> 3;
      int db  = slot & 7;
      vreg[j] = *(const v8h*)(Vp + (size_t)(kc + key) * INNER_ + db * 8);
    }

    // Prefetch next chunk's K and V rows into cache (gfx1250 prefetch path)
    if (kc + 64 < M_) {
      int pr = kc + 64 + lane;                // 32 rows per wave, x2
      __builtin_prefetch(Kp + (size_t)pr * INNER_, 0, 3);
      __builtin_prefetch(Kp + (size_t)(pr + 32) * INNER_, 0, 3);
      __builtin_prefetch(Vp + (size_t)pr * INNER_, 0, 3);
      __builtin_prefetch(Vp + (size_t)(pr + 32) * INNER_, 0, 3);
    }

    // --- Stage 2: S = Q*K^T ; all 8 K fragments loaded before the WMMAs ---
    v16h kbf[8];
#pragma unroll
    for (int t = 0; t < 4; ++t) {
      const _Float16* kp = Kp + (size_t)(kc + t * 16) * INNER_;
      kbf[2 * t]     = load_fragB(kp, INNER_);
      kbf[2 * t + 1] = load_fragB(kp + 32, INNER_);
    }
    v8f s[4];
#pragma unroll
    for (int t = 0; t < 4; ++t) {
      v8f a = (v8f){};
      a = wmma_f16(qa0, kbf[2 * t], a);
      a = wmma_f16(qa1, kbf[2 * t + 1], a);
#pragma unroll
      for (int r = 0; r < 8; ++r) a[r] *= scale;
      s[t] = a;
    }

    // --- Stage 3: online softmax (hides the V loads issued in stage 1) ----
    float alpha[8];
#pragma unroll
    for (int r = 0; r < 8; ++r) {
      float mx = fmaxf(fmaxf(s[0][r], s[1][r]), fmaxf(s[2][r], s[3][r]));
#pragma unroll
      for (int off = 1; off < 16; off <<= 1)
        mx = fmaxf(mx, __shfl_xor(mx, off, 32));
      float nm = fmaxf(mrun[r], mx);
      alpha[r] = __expf(mrun[r] - nm);
      mrun[r]  = nm;
      float rs = 0.f;
#pragma unroll
      for (int t = 0; t < 4; ++t) {
        float e = __expf(s[t][r] - nm);
        s[t][r] = e;
        rs += e;
      }
#pragma unroll
      for (int off = 1; off < 16; off <<= 1)
        rs += __shfl_xor(rs, off, 32);
      lrun[r] = lrun[r] * alpha[r] + rs;
    }
#pragma unroll
    for (int dt = 0; dt < 4; ++dt)
#pragma unroll
      for (int r = 0; r < 8; ++r) o[dt][r] *= alpha[r];

    // --- Stage 4: spill V regs transposed into LDS, then barrier ----------
#pragma unroll
    for (int j = 0; j < 2; ++j) {
      int slot = threadIdx.x + j * 256;
      int key = slot >> 3;
      int db  = slot & 7;
#pragma unroll
      for (int e = 0; e < 8; ++e)
        sVt[(db * 8 + e) * 72 + key] = vreg[j][e];
    }
    __syncthreads();

    // --- Stage 5: O += P*V (P via wave-private LDS re-layout) -------------
    _Float16* pw = &sP[wave][0];
#pragma unroll
    for (int t = 0; t < 4; ++t)
#pragma unroll
      for (int r = 0; r < 8; ++r)
        pw[(r + 8 * hs) * 72 + t * 16 + cn] = (_Float16)s[t][r];

    v16h pa0 = load_fragA(pw, 72);
    v16h pa1 = load_fragA(pw + 32, 72);
#pragma unroll
    for (int dt = 0; dt < 4; ++dt) {
      v16h vb0 = load_fragB(sVt + dt * 16 * 72, 72);
      v16h vb1 = load_fragB(sVt + dt * 16 * 72 + 32, 72);
      o[dt] = wmma_f16(pa0, vb0, o[dt]);
      o[dt] = wmma_f16(pa1, vb1, o[dt]);
    }
    __syncthreads();   // protect sVt before next chunk overwrite
  }

  // Normalize and store O (f16) back into [token, h*64+d] layout
  _Float16* op = Oh + qbase + (size_t)qrow0 * INNER_ + h * DH_;
  float inv[8];
#pragma unroll
  for (int r = 0; r < 8; ++r) inv[r] = 1.f / lrun[r];
#pragma unroll
  for (int dt = 0; dt < 4; ++dt)
#pragma unroll
    for (int r = 0; r < 8; ++r)
      op[(size_t)(r + 8 * hs) * INNER_ + dt * 16 + cn] =
          (_Float16)(o[dt][r] * inv[r]);
}

// ---------------------------------------------------------------------------
// Host launch
// ---------------------------------------------------------------------------
extern "C" void kernel_launch(void* const* d_in, const int* in_sizes, int n_in,
                              void* d_out, int out_size, void* d_ws,
                              size_t ws_size, hipStream_t stream) {
  const float* x   = (const float*)d_in[0];
  const float* ctx = (const float*)d_in[1];
  const float* Wq  = (const float*)d_in[2];
  const float* Wk  = (const float*)d_in[3];
  const float* Wv  = (const float*)d_in[4];
  const float* Wo  = (const float*)d_in[5];
  const float* bo  = (const float*)d_in[6];
  float* out = (float*)d_out;

  char* ws = (char*)d_ws;
  size_t off = 0;
  auto alloc_h = [&](size_t nhalf) {
    _Float16* p = (_Float16*)(ws + off);
    off += nhalf * sizeof(_Float16);
    return p;
  };
  const size_t ROWS = (size_t)B_ * N_;          // 4096
  _Float16* xh  = alloc_h(ROWS * INNER_);       // x   in f16
  _Float16* ch  = alloc_h(ROWS * INNER_);       // ctx in f16
  _Float16* Wqt = alloc_h((size_t)INNER_ * INNER_);
  _Float16* Wkt = alloc_h((size_t)INNER_ * INNER_);
  _Float16* Wvt = alloc_h((size_t)INNER_ * INNER_);
  _Float16* Wot = alloc_h((size_t)INNER_ * INNER_);
  _Float16* Qh  = alloc_h(ROWS * INNER_);
  _Float16* Kh  = alloc_h(ROWS * INNER_);
  _Float16* Vh  = alloc_h(ROWS * INNER_);
  _Float16* Ohh = alloc_h(ROWS * INNER_);
  (void)ws_size; (void)in_sizes; (void)n_in; (void)out_size;

  const int NELEM = (int)(ROWS * INNER_);       // 4,194,304
  cvt_f16_kernel<<<(NELEM / 8) / 256, 256, 0, stream>>>(x, xh, NELEM / 8);
  cvt_f16_kernel<<<(NELEM / 8) / 256, 256, 0, stream>>>(ctx, ch, NELEM / 8);
  transpose_cvt_kernel<<<(1024 * 1024) / 256, 256, 0, stream>>>(Wq, Wqt);
  transpose_cvt_kernel<<<(1024 * 1024) / 256, 256, 0, stream>>>(Wk, Wkt);
  transpose_cvt_kernel<<<(1024 * 1024) / 256, 256, 0, stream>>>(Wv, Wvt);
  transpose_cvt_kernel<<<(1024 * 1024) / 256, 256, 0, stream>>>(Wo, Wot);

  dim3 gg(32, 8);                               // 4096/128 x 1024/128
  gemm_wmma_kernel<false><<<gg, 256, 0, stream>>>(xh, Wqt, Qh, nullptr,
                                                  nullptr, 4096, 1024, 1024);
  gemm_wmma_kernel<false><<<gg, 256, 0, stream>>>(ch, Wkt, Kh, nullptr,
                                                  nullptr, 4096, 1024, 1024);
  gemm_wmma_kernel<false><<<gg, 256, 0, stream>>>(ch, Wvt, Vh, nullptr,
                                                  nullptr, 4096, 1024, 1024);

  attn_wmma_kernel<<<dim3(N_ / 128, HEADS_, B_), 256, 0, stream>>>(Qh, Kh, Vh,
                                                                   Ohh);

  gemm_wmma_kernel<true><<<gg, 256, 0, stream>>>(Ohh, Wot, nullptr, out, bo,
                                                 4096, 1024, 1024);
}
